// DeepSeekMoE_66795331387842
// MI455X (gfx1250) — matrix operand order repaired and checked by reading, not compiled, not run on hardware
//
#include <hip/hip_runtime.h>
#include <hip/hip_bf16.h>

typedef __attribute__((ext_vector_type(16))) __bf16 v16bf;
typedef __attribute__((ext_vector_type(8)))  __bf16 v8bf;
typedef __attribute__((ext_vector_type(8)))  float  v8f;
typedef __attribute__((ext_vector_type(4)))  float  v4f;

#define S_DIM 1024
#define B_DIM 2
#define H_DIM 1024
#define E_DIM 8
#define T_DIM 2048        // B*S
#define FFN   1408
#define LDS_P 40          // padded row stride in bf16 elems (80B, 16B-aligned)

#if defined(__has_builtin)
#if __has_builtin(__builtin_amdgcn_sched_group_barrier)
#define SCHED_GROUP(mask, n) __builtin_amdgcn_sched_group_barrier((mask), (n), 0)
#endif
#endif
#ifndef SCHED_GROUP
#define SCHED_GROUP(mask, n)
#endif

// ---------------------------------------------------------------------------
// Kernel 1: gating. One wave per token. 8 scores -> softmax -> top2 -> append
// (token, normalized weight) into per-expert gather lists via atomics.
// ---------------------------------------------------------------------------
__global__ __launch_bounds__(256) void moe_gate_kernel(
    const float* __restrict__ hid, const float* __restrict__ gw,
    unsigned* __restrict__ counts, int* __restrict__ idx_list,
    float* __restrict__ wgt_list)
{
    int wave = threadIdx.x >> 5;
    int lane = threadIdx.x & 31;
    int t = blockIdx.x * 8 + wave;
    if (t >= T_DIM) return;

    int s = t % S_DIM, b = t / S_DIM;
    const float* xr = hid + ((size_t)s * B_DIM + b) * H_DIM;

    float acc[E_DIM];
#pragma unroll
    for (int e = 0; e < E_DIM; ++e) acc[e] = 0.0f;

    for (int h = lane; h < H_DIM; h += 32) {
        float xv = xr[h];
#pragma unroll
        for (int e = 0; e < E_DIM; ++e) acc[e] += xv * gw[e * H_DIM + h];
    }
#pragma unroll
    for (int off = 16; off > 0; off >>= 1) {
#pragma unroll
        for (int e = 0; e < E_DIM; ++e)
            acc[e] += __shfl_xor(acc[e], off, 32);
    }

    if (lane == 0) {
        float m = acc[0];
#pragma unroll
        for (int e = 1; e < E_DIM; ++e) m = fmaxf(m, acc[e]);
        float p[E_DIM], sum = 0.0f;
#pragma unroll
        for (int e = 0; e < E_DIM; ++e) { p[e] = __expf(acc[e] - m); sum += p[e]; }
        float inv = 1.0f / sum;
#pragma unroll
        for (int e = 0; e < E_DIM; ++e) p[e] *= inv;

        int i0 = 0;
#pragma unroll
        for (int e = 1; e < E_DIM; ++e) if (p[e] > p[i0]) i0 = e;
        int i1 = (i0 == 0) ? 1 : 0;
#pragma unroll
        for (int e = 0; e < E_DIM; ++e)
            if (e != i0 && p[e] > p[i1]) i1 = e;

        float denom = p[i0] + p[i1] + 1e-20f;
        float w0 = p[i0] / denom;
        float w1 = p[i1] / denom;

        unsigned s0 = atomicAdd(&counts[i0], 1u);
        idx_list[i0 * T_DIM + s0] = t;
        wgt_list[i0 * T_DIM + s0] = w0;
        unsigned s1 = atomicAdd(&counts[i1], 1u);
        idx_list[i1 * T_DIM + s1] = t;
        wgt_list[i1 * T_DIM + s1] = w1;
    }
}

// ---------------------------------------------------------------------------
// Kernel 2: exclusive prefix over 8 expert counts -> row bases in act buffer.
// ---------------------------------------------------------------------------
__global__ void moe_prefix_kernel(const unsigned* __restrict__ counts,
                                  unsigned* __restrict__ bases)
{
    if (threadIdx.x == 0) {
        unsigned a = 0;
#pragma unroll
        for (int e = 0; e < E_DIM; ++e) { bases[e] = a; a += counts[e]; }
    }
}

// ---- small helpers --------------------------------------------------------
__device__ __forceinline__ v8bf pack_bf8(v4f a, v4f b)
{
    v8bf r;
#pragma unroll
    for (int i = 0; i < 4; ++i) { r[i] = (__bf16)a[i]; r[4 + i] = (__bf16)b[i]; }
    return r;
}

// ---- WMMA fragment helpers (16-bit A/B layouts, ISA 7.12.2) ---------------
__device__ __forceinline__ v16bf load_frag_a(const __bf16* row, int hf)
{
    // A 16x32: lane m=lane&15, half=lane>>4. v0..3 hold K = half*8+[0..7],
    // v4..7 hold K = 16 + half*8 + [0..7]  -> two contiguous 16B chunks.
    v16bf a;
    v8bf lo = *(const v8bf*)(row + hf * 8);
    v8bf hi = *(const v8bf*)(row + 16 + hf * 8);
#pragma unroll
    for (int i = 0; i < 8; ++i) { a[i] = lo[i]; a[8 + i] = hi[i]; }
    return a;
}

__device__ __forceinline__ v16bf load_frag_b(const __bf16* col, int hf)
{
    // B 32x16 (stored transposed in LDS [n][k]): lane n=lane&15, half=lane>>4.
    // v0..3 hold K = half*8+[0..7], v4..7 hold K = 16 + half*8 + [0..7].
    v16bf bfr;
    v8bf lo = *(const v8bf*)(col + hf * 8);
    v8bf hi = *(const v8bf*)(col + 16 + hf * 8);
#pragma unroll
    for (int i = 0; i < 8; ++i) { bfr[i] = lo[i]; bfr[8 + i] = hi[i]; }
    return bfr;
}

// ---------------------------------------------------------------------------
// Kernel 3: grouped GEMM1 + SwiGLU.  h = x_gathered @ w1[e]; act = silu(g)*u.
// 64x64 tile per WG (128 threads = 4 waves), double-buffered LDS pipeline,
// branch-free loop body, sched-group pipeline: VMEM -> 18 DS reads -> 8 WMMA.
// ---------------------------------------------------------------------------
__global__ __launch_bounds__(128, 1) void moe_gemm1_kernel(
    const float* __restrict__ hid, const float* __restrict__ w1,
    const unsigned* __restrict__ counts, const unsigned* __restrict__ bases,
    const int* __restrict__ idx_list, unsigned short* __restrict__ act_raw)
{
    __bf16* act = (__bf16*)act_raw;
    const int e = blockIdx.z;
    const unsigned Ne = counts[e];
    const int trow0 = blockIdx.y * 64;
    if ((unsigned)trow0 >= Ne) return;
    const int f0 = blockIdx.x * 64;

    const float* w1e = w1 + (size_t)e * H_DIM * (2 * FFN);
    const int* idx = idx_list + e * T_DIM;
    const unsigned base = bases[e];

    __shared__ __align__(16) __bf16 As[2][64][LDS_P];
    __shared__ __align__(16) __bf16 Bs[2][2][64][LDS_P];

    const int tid  = threadIdx.x;
    const int lane = tid & 31;
    const int wave = tid >> 5;
    const int hf   = lane >> 4;
    const int ln   = lane & 15;
    const int wrow = wave * 16;

    // staging coordinates
    const int srow = tid >> 1;              // A: row 0..63
    const int skh  = (tid & 1) * 16;        // A: k sub-block 0 / 16
    const int sc4  = (tid & 15) * 4;        // B: col group (4 cols)
    const int skk0 = tid >> 4;              // B: k start 0..7 (stride 8)

    // Gather row, clamped: OOB rows load a valid row; the garbage only lands
    // in accumulator rows >= Ne which the epilogue masks out.
    const int arow = trow0 + srow;
    const int crow = ((unsigned)arow < Ne) ? arow : (int)(Ne - 1);
    const int ttok = idx[crow];
    const float* asrc =
        hid + ((size_t)(ttok % S_DIM) * B_DIM + (ttok / S_DIM)) * H_DIM + skh;

    v8f accg[4] = {};
    v8f accu[4] = {};

    v4f ra[4], rg[4], ru[4];

    // global -> registers for K block starting at k0 (branch-free)
    auto stage_load = [&](int k0) {
#pragma unroll
        for (int j = 0; j < 4; ++j)
            ra[j] = *(const v4f*)(asrc + k0 + j * 4);
#pragma unroll
        for (int i = 0; i < 4; ++i) {
            int kk = skk0 + i * 8;
            size_t roff = (size_t)(k0 + kk) * (2 * FFN);
            rg[i] = *(const v4f*)(w1e + roff + f0 + sc4);
            ru[i] = *(const v4f*)(w1e + roff + FFN + f0 + sc4);
        }
    };
    // registers -> LDS buffer p
    auto stage_store = [&](int p) {
        *(v8bf*)(&As[p][srow][skh])     = pack_bf8(ra[0], ra[1]);
        *(v8bf*)(&As[p][srow][skh + 8]) = pack_bf8(ra[2], ra[3]);
#pragma unroll
        for (int i = 0; i < 4; ++i) {
            int kk = skk0 + i * 8;
#pragma unroll
            for (int j = 0; j < 4; ++j) {
                Bs[p][0][sc4 + j][kk] = (__bf16)rg[i][j];
                Bs[p][1][sc4 + j][kk] = (__bf16)ru[i][j];
            }
        }
    };

    stage_load(0);
    stage_store(0);

    const int KSTEPS = H_DIM / 32;          // 32
    for (int step = 0; step < KSTEPS; ++step) {
        __syncthreads();
        const int cur = step & 1;
        // last iteration redundantly re-stages block 0 into the dead buffer
        const int knext = (step + 1 < KSTEPS) ? (step + 1) * 32 : 0;

        stage_load(knext);

        // preload all fragments into distinct regs, then burst the WMMAs
        v16bf a = load_frag_a(&As[cur][wrow + ln][0], hf);
        v16bf bg[4], bu[4];
#pragma unroll
        for (int sub = 0; sub < 4; ++sub) {
            bg[sub] = load_frag_b(&Bs[cur][0][sub * 16 + ln][0], hf);
            bu[sub] = load_frag_b(&Bs[cur][1][sub * 16 + ln][0], hf);
        }
#pragma unroll
        for (int sub = 0; sub < 4; ++sub) {
            accg[sub] = __builtin_amdgcn_wmma_f32_16x16x32_bf16(
                false, a, false, bg[sub], (short)0, accg[sub], false, false);
            accu[sub] = __builtin_amdgcn_wmma_f32_16x16x32_bf16(
                false, a, false, bu[sub], (short)0, accu[sub], false, false);
        }

        stage_store(cur ^ 1);

        // pipeline: global loads first, then all LDS frag reads, then the
        // WMMA burst, then LDS stores for the next step.
        SCHED_GROUP(0x020, 12);   // VMEM reads
        SCHED_GROUP(0x100, 18);   // DS reads (A:2 + B:16)
        SCHED_GROUP(0x008, 8);    // WMMA
        SCHED_GROUP(0x200, 34);   // DS writes
    }

    // Epilogue: act = silu(g) * u, bf16 to workspace (gathered row-major [*, FFN])
    const unsigned rbase = (unsigned)(trow0 + wrow + hf * 8);
#pragma unroll
    for (int sub = 0; sub < 4; ++sub) {
#pragma unroll
        for (int r = 0; r < 8; ++r) {
            unsigned row = rbase + r;
            if (row < Ne) {
                float g = accg[sub][r];
                float u = accu[sub][r];
                float sg = g / (1.0f + __expf(-g));
                act[(size_t)(base + row) * FFN + f0 + sub * 16 + ln] = (__bf16)(sg * u);
            }
        }
    }
}

// ---------------------------------------------------------------------------
// Kernel 4: grouped GEMM2 + weighted scatter.  y = act @ w2[e]; out += w * y.
// ---------------------------------------------------------------------------
__global__ __launch_bounds__(128, 1) void moe_gemm2_kernel(
    const unsigned short* __restrict__ act_raw, const float* __restrict__ w2,
    const unsigned* __restrict__ counts, const unsigned* __restrict__ bases,
    const int* __restrict__ idx_list, const float* __restrict__ wgt_list,
    float* __restrict__ out)
{
    const __bf16* act = (const __bf16*)act_raw;
    const int e = blockIdx.z;
    const unsigned Ne = counts[e];
    const int trow0 = blockIdx.y * 64;
    if ((unsigned)trow0 >= Ne) return;
    const int n0 = blockIdx.x * 64;

    const float* w2e = w2 + (size_t)e * FFN * H_DIM;
    const unsigned base = bases[e];

    __shared__ __align__(16) __bf16 As[2][64][LDS_P];
    __shared__ __align__(16) __bf16 Bs[2][64][LDS_P];

    const int tid  = threadIdx.x;
    const int lane = tid & 31;
    const int wave = tid >> 5;
    const int hf   = lane >> 4;
    const int ln   = lane & 15;
    const int wrow = wave * 16;

    const int srow = tid >> 1;
    const int skh  = (tid & 1) * 16;
    const int sc4  = (tid & 15) * 4;
    const int skk0 = tid >> 4;

    const int arow = trow0 + srow;
    const int crow = ((unsigned)arow < Ne) ? arow : (int)(Ne - 1);
    const __bf16* asrc = act + (size_t)(base + crow) * FFN + skh;

    v8f acc[4] = {};

    v8bf ralo, rahi;
    v4f rb[4];

    auto stage_load = [&](int k0) {
        ralo = *(const v8bf*)(asrc + k0);
        rahi = *(const v8bf*)(asrc + k0 + 8);
#pragma unroll
        for (int i = 0; i < 4; ++i) {
            int kk = skk0 + i * 8;
            rb[i] = *(const v4f*)(w2e + (size_t)(k0 + kk) * H_DIM + n0 + sc4);
        }
    };
    auto stage_store = [&](int p) {
        *(v8bf*)(&As[p][srow][skh])     = ralo;
        *(v8bf*)(&As[p][srow][skh + 8]) = rahi;
#pragma unroll
        for (int i = 0; i < 4; ++i) {
            int kk = skk0 + i * 8;
#pragma unroll
            for (int j = 0; j < 4; ++j)
                Bs[p][sc4 + j][kk] = (__bf16)rb[i][j];
        }
    };

    stage_load(0);
    stage_store(0);

    const int KSTEPS = FFN / 32;            // 44
    for (int step = 0; step < KSTEPS; ++step) {
        __syncthreads();
        const int cur = step & 1;
        const int knext = (step + 1 < KSTEPS) ? (step + 1) * 32 : 0;

        stage_load(knext);

        v16bf a = load_frag_a(&As[cur][wrow + ln][0], hf);
        v16bf bf4[4];
#pragma unroll
        for (int sub = 0; sub < 4; ++sub)
            bf4[sub] = load_frag_b(&Bs[cur][sub * 16 + ln][0], hf);
#pragma unroll
        for (int sub = 0; sub < 4; ++sub)
            acc[sub] = __builtin_amdgcn_wmma_f32_16x16x32_bf16(
                false, a, false, bf4[sub], (short)0, acc[sub], false, false);

        stage_store(cur ^ 1);

        SCHED_GROUP(0x020, 6);    // VMEM reads
        SCHED_GROUP(0x100, 10);   // DS reads (A:2 + B:8)
        SCHED_GROUP(0x008, 4);    // WMMA
        SCHED_GROUP(0x200, 18);   // DS writes
    }

    // Epilogue: scale by routing weight, scatter-add into out[S,B,H]
    const unsigned rbase = (unsigned)(trow0 + wrow + hf * 8);
#pragma unroll
    for (int sub = 0; sub < 4; ++sub) {
#pragma unroll
        for (int r = 0; r < 8; ++r) {
            unsigned row = rbase + r;
            if (row < Ne) {
                float w = wgt_list[e * T_DIM + row];
                int t = idx_list[e * T_DIM + row];
                size_t o = ((size_t)(t % S_DIM) * B_DIM + (t / S_DIM)) * H_DIM
                           + n0 + sub * 16 + ln;
                atomicAdd(&out[o], w * acc[sub][r]);
            }
        }
    }
}

// ---------------------------------------------------------------------------
// Host launcher. Workspace layout (bytes):
//   [0,32)        counts[8]
//   [32,64)       bases[8]
//   [64,65600)    idx_list[E*T] (int)
//   [65600,131136) wgt_list[E*T] (float)
//   [131136, +2T*FFN*2) act (bf16, 4096 x 1408)   ~11.5 MB
// ---------------------------------------------------------------------------
extern "C" void kernel_launch(void* const* d_in, const int* in_sizes, int n_in,
                              void* d_out, int out_size, void* d_ws, size_t ws_size,
                              hipStream_t stream)
{
    const float* hid = (const float*)d_in[0];
    const float* gw  = (const float*)d_in[1];
    const float* w1  = (const float*)d_in[2];
    const float* w2  = (const float*)d_in[3];
    float* out = (float*)d_out;

    char* ws = (char*)d_ws;
    unsigned* counts        = (unsigned*)(ws + 0);
    unsigned* bases         = (unsigned*)(ws + 32);
    int* idx_list           = (int*)(ws + 64);
    float* wgt_list         = (float*)(ws + 64 + E_DIM * T_DIM * 4);
    unsigned short* act_raw = (unsigned short*)(ws + 64 + 2 * E_DIM * T_DIM * 4);

    hipMemsetAsync(counts, 0, 32, stream);
    hipMemsetAsync(d_out, 0, (size_t)out_size * sizeof(float), stream);

    moe_gate_kernel<<<T_DIM / 8, 256, 0, stream>>>(hid, gw, counts, idx_list, wgt_list);
    moe_prefix_kernel<<<1, 32, 0, stream>>>(counts, bases);

    dim3 g1(FFN / 64, T_DIM / 64, E_DIM);   // 22 x 32 x 8
    moe_gemm1_kernel<<<g1, 128, 0, stream>>>(hid, w1, counts, bases, idx_list, act_raw);

    dim3 g2(H_DIM / 64, T_DIM / 64, E_DIM); // 16 x 32 x 8
    moe_gemm2_kernel<<<g2, 128, 0, stream>>>(act_raw, w2, counts, bases, idx_list,
                                             wgt_list, out);
}
